// Attention_37434934952112
// MI455X (gfx1250) — matrix operand order repaired and checked by reading, not compiled, not run on hardware
//
#include <hip/hip_runtime.h>
#include <math.h>

typedef _Float16 h16;
typedef __attribute__((ext_vector_type(16))) _Float16 v16h;
typedef __attribute__((ext_vector_type(8)))  float    v8f;
typedef __attribute__((__vector_size__(16))) int      vi4;

#define B_   128
#define N_   197
#define C_   768
#define H_   12
#define HD_  64
#define NP_  224      // padded seq len (multiple of 32)
#define NPAT 14
#define P_   12
#define SCALE 0.125f  // hd^-0.5

// ---------------- async global->LDS staging (guarded CDNA5 path) ------------
#if defined(__has_builtin)
#if __has_builtin(__builtin_amdgcn_global_load_async_to_lds_b128)
#define TRY_ASYNC 1
#endif
#endif
#ifndef TRY_ASYNC
#define TRY_ASYNC 0
#endif

__device__ __forceinline__ void copy16_g2l(h16* l, const h16* g) {
#if TRY_ASYNC
  __builtin_amdgcn_global_load_async_to_lds_b128(
      (__attribute__((address_space(1))) vi4*)g,
      (__attribute__((address_space(3))) vi4*)l, 0, 0);
#else
  *(uint4*)l = *(const uint4*)g;
#endif
}

__device__ __forceinline__ void wait_async_copies() {
#if TRY_ASYNC
#if __has_builtin(__builtin_amdgcn_s_wait_asynccnt)
  __builtin_amdgcn_s_wait_asynccnt(0);
#else
  asm volatile("s_wait_asynccnt 0" ::: "memory");
#endif
#endif
}

// ---------------- WMMA fragment loaders (layouts per CDNA5 ISA) --

// A: 16x32 f16 from row-major source, leading dim lda.
__device__ __forceinline__ v16h load_a_rm(const h16* __restrict__ A, int lda, int lane) {
  v16h a;
  int r  = lane & 15;
  int ko = (lane >> 4) << 3;          // 0 or 8
  const h16* p = A + (size_t)r * lda + ko;
#pragma unroll
  for (int e = 0; e < 8; ++e) {
    a[e]     = p[e];                  // K = ko + e
    a[8 + e] = p[16 + e];             // K = 16 + ko + e
  }
  return a;
}

// B: 32x16 where B[kk][col] = S[col*lds + kk] (K contiguous per column)
__device__ __forceinline__ v16h load_b_tr(const h16* __restrict__ S, int lds_, int lane) {
  v16h b;
  int col = lane & 15;
  int ko  = (lane >> 4) << 3;         // 0 or 8
  const h16* p = S + (size_t)col * lds_ + ko;
#pragma unroll
  for (int e = 0; e < 8; ++e) { b[e] = p[e]; b[8 + e] = p[16 + e]; }
  return b;
}

#define WMMA_F16(a, b, c) \
  __builtin_amdgcn_wmma_f32_16x16x32_f16(false, (a), false, (b), (short)0, (c), false, false)

// ---------------- elementwise cast -----------------------------------------
__global__ void cast_f16_kernel(const float* __restrict__ in, h16* __restrict__ out, int n) {
  int i = blockIdx.x * blockDim.x + threadIdx.x;
  int stride = gridDim.x * blockDim.x;
  for (; i < n; i += stride) out[i] = (h16)in[i];
}

// ---------------- transpose + cast: in [K][N] f32 -> out [N][K] f16 ---------
__global__ void castT_f16_kernel(const float* __restrict__ in, h16* __restrict__ out,
                                 int K, int N) {
  int idx = blockIdx.x * blockDim.x + threadIdx.x;
  int total = K * N;
  if (idx < total) {
    int n = idx / K, k = idx % K;     // idx linear over output [N][K]
    out[idx] = (h16)in[(size_t)k * N + n];
  }
}

// ---------------- qkv GEMM: 128x128 tiles, 8 waves, LDS staged --------------
// A = xh [M][768];  B = wqt [2304][768] (transposed weight, K contiguous)
__global__ void gemm_qkv_kernel(const h16* __restrict__ xh, const h16* __restrict__ wqt,
                                h16* __restrict__ qh, h16* __restrict__ kh,
                                h16* __restrict__ vh, h16* __restrict__ vt) {
  __shared__ __align__(16) h16 As[128 * 32];   // A tile [128 rows][32 k]
  __shared__ __align__(16) h16 Bts[128 * 32];  // B tile [128 cols][32 k]
  int tid = threadIdx.x, lane = tid & 31, w = tid >> 5;
  int wm = (w >> 2) * 64;          // 0 / 64
  int wn = (w & 3) * 32;           // 0 / 32 / 64 / 96
  int bm = blockIdx.x, bn = blockIdx.y;
  const h16* Ab = xh + (size_t)bm * 128 * C_;
  const h16* Bb = wqt + (size_t)bn * 128 * C_;
  v8f acc[4][2] = {};
#pragma unroll 1
  for (int k0 = 0; k0 < C_; k0 += 32) {
#pragma unroll
    for (int s = tid; s < 512; s += 256) {      // A: 128x32 halves
      int row = s >> 2, co = (s & 3) * 8;
      copy16_g2l(As + row * 32 + co, Ab + (size_t)row * C_ + k0 + co);
    }
#pragma unroll
    for (int s = tid; s < 512; s += 256) {      // B^T: 128x32 halves
      int row = s >> 2, co = (s & 3) * 8;
      copy16_g2l(Bts + row * 32 + co, Bb + (size_t)row * C_ + k0 + co);
    }
    wait_async_copies();
    __syncthreads();
    v16h bfr[2];
#pragma unroll
    for (int nt = 0; nt < 2; ++nt) bfr[nt] = load_b_tr(Bts + (wn + nt * 16) * 32, 32, lane);
#pragma unroll
    for (int mt = 0; mt < 4; ++mt) {
      v16h a = load_a_rm(As + (wm + mt * 16) * 32, 32, lane);
#pragma unroll
      for (int nt = 0; nt < 2; ++nt) acc[mt][nt] = WMMA_F16(a, bfr[nt], acc[mt][nt]);
    }
    __syncthreads();
  }
  int col = lane & 15, rb = (lane >> 4) * 8;
#pragma unroll
  for (int nt = 0; nt < 2; ++nt) {
    int gj = bn * 128 + wn + nt * 16 + col;
    int which = gj / C_, cc = gj % C_;
    int h = cc >> 6, d = cc & 63;
    h16* dst = (which == 0) ? qh : ((which == 1) ? kh : vh);
    float sc = (which == 0) ? SCALE : 1.0f;
#pragma unroll
    for (int i = 0; i < 8; ++i) {
      for (int mt = 0; mt < 4; ++mt) {
        int gi = bm * 128 + wm + mt * 16 + rb + i;
        int b_ = gi / N_, n = gi % N_;
        h16 val = (h16)(acc[mt][nt][i] * sc);
        dst[(((size_t)b_ * H_ + h) * NP_ + n) * HD_ + d] = val;
        if (which == 2)  // extra d-major copy of v for the w@v GEMM
          vt[(((size_t)b_ * H_ + h) * HD_ + d) * NP_ + n] = val;
      }
    }
  }
}

// ---------------- per-head mean + global mean of v --------------------------
__global__ void stats_kernel(const h16* __restrict__ vh, float* __restrict__ vmh,
                             float* __restrict__ vmi) {
  int b = blockIdx.x, tid = threadIdx.x;
  __shared__ float sm[H_ * HD_];
  for (int idx = tid; idx < H_ * HD_; idx += blockDim.x) {
    int h = idx >> 6, d = idx & 63;
    const h16* base = vh + (((size_t)b * H_ + h) * NP_) * HD_ + d;
    float s = 0.f;
    for (int t = 1; t < N_; ++t) s += (float)base[(size_t)t * HD_];
    float m = s * (1.0f / (float)(N_ - 1));
    sm[idx] = m;
    vmh[((size_t)b * H_ + h) * HD_ + d] = m;
  }
  __syncthreads();
  if (tid < HD_) {
    float s = 0.f;
    for (int h = 0; h < H_; ++h) s += sm[h * HD_ + tid];
    vmi[(size_t)b * HD_ + tid] = s * (1.0f / (float)H_);
  }
}

// ---------------- per-patch distance from global mean -----------------------
__global__ void dist_kernel(const h16* __restrict__ vh, const float* __restrict__ vmi,
                            float* __restrict__ dist) {
  int b = blockIdx.x, p = threadIdx.x;
  __shared__ float mu[HD_];
  for (int d = threadIdx.x; d < HD_; d += blockDim.x) mu[d] = vmi[(size_t)b * HD_ + d];
  __syncthreads();
  if (p < NPAT * NPAT) {
    float acc = 0.f;
    for (int h = 0; h < H_; ++h) {
      const h16* row = vh + (((size_t)b * H_ + h) * NP_ + (p + 1)) * HD_;
      float ss = 0.f;
      for (int d = 0; d < HD_; ++d) { float df = (float)row[d] - mu[d]; ss += df * df; }
      acc += sqrtf(ss);
    }
    dist[b * (NPAT * NPAT) + p] = acc * (1.0f / (float)H_);
  }
}

// ---------------- 3x3 mean-pool + argmax + token mask ------------------------
__global__ void mask_kernel(const float* __restrict__ dist, unsigned char* __restrict__ mask) {
  int b = blockIdx.x, tid = threadIdx.x;
  __shared__ float ds[NPAT * NPAT];
  __shared__ float pool[P_ * P_];
  __shared__ int best;
  for (int i = tid; i < NPAT * NPAT; i += blockDim.x) ds[i] = dist[b * (NPAT * NPAT) + i];
  __syncthreads();
  if (tid < P_ * P_) {
    int i = tid / P_, j = tid % P_;
    float s = 0.f;
    for (int a = 0; a < 3; ++a)
      for (int cc = 0; cc < 3; ++cc) s += ds[(i + a) * NPAT + (j + cc)];
    pool[tid] = s * (1.0f / 9.0f);
  }
  __syncthreads();
  if (tid == 0) {  // first-max tie-break to match jnp.argmax
    int bi = 0; float bv = pool[0];
    for (int t = 1; t < P_ * P_; ++t) if (pool[t] > bv) { bv = pool[t]; bi = t; }
    best = bi;
  }
  __syncthreads();
  if (tid < NP_) mask[b * NP_ + tid] = 0;
  __syncthreads();
  if (tid < 9) {
    int i = best / P_, j = best % P_;
    int a = tid / 3, cc = tid % 3;
    int token = NPAT * (i + a) + (j + cc) + 1;
    mask[b * NP_ + token] = 1;
  }
}

// ---------------- scores: one block per (b,h), K-tile in LDS ----------------
__global__ void gemm_scores_kernel(const h16* __restrict__ qh, const h16* __restrict__ kh,
                                   h16* __restrict__ wh) {
  __shared__ __align__(16) h16 Ks[NP_ * HD_];  // 224x64 = 28 KB
  int tid = threadIdx.x, lane = tid & 31, w = tid >> 5;   // w = 0..13
  int bh = blockIdx.x;
  const h16* Q = qh + (size_t)bh * NP_ * HD_;
  const h16* K = kh + (size_t)bh * NP_ * HD_;
#pragma unroll
  for (int s = tid; s < (NP_ * HD_) / 8; s += 448)
    copy16_g2l(Ks + s * 8, K + (size_t)s * 8);
  wait_async_copies();
  __syncthreads();
  v8f acc[14];
#pragma unroll
  for (int t = 0; t < 14; ++t) acc[t] = (v8f){};
#pragma unroll
  for (int k0 = 0; k0 < HD_; k0 += 32) {
    v16h a = load_a_rm(Q + (size_t)w * 16 * HD_ + k0, HD_, lane);
#pragma unroll
    for (int tn = 0; tn < 14; ++tn) {
      v16h b = load_b_tr(Ks + tn * 16 * HD_ + k0, HD_, lane);
      acc[tn] = WMMA_F16(a, b, acc[tn]);
    }
  }
  int col = lane & 15, rb = (lane >> 4) * 8;
  h16* W = wh + (size_t)bh * NP_ * NP_;
#pragma unroll
  for (int tn = 0; tn < 14; ++tn)
#pragma unroll
    for (int i = 0; i < 8; ++i)
      W[(size_t)(w * 16 + rb + i) * NP_ + tn * 16 + col] = (h16)acc[tn][i];
}

// ---------------- softmax over 197 keys + masked-key substitution -----------
__global__ void softmax_mask_kernel(h16* __restrict__ wh, const unsigned char* __restrict__ mask) {
  int n = blockIdx.x, h = blockIdx.y, b = blockIdx.z;
  int j = threadIdx.x;
  int bh = b * H_ + h;
  h16* row = wh + ((size_t)bh * NP_ + n) * NP_;
  __shared__ float red[256];
  float s = (j < N_) ? (float)row[j] : -1e30f;
  red[j] = s;
  __syncthreads();
  for (int o = 128; o > 0; o >>= 1) { if (j < o) red[j] = fmaxf(red[j], red[j + o]); __syncthreads(); }
  float m = red[0];
  __syncthreads();
  float e = (j < N_) ? __expf(s - m) : 0.f;
  red[j] = e;
  __syncthreads();
  for (int o = 128; o > 0; o >>= 1) { if (j < o) red[j] += red[j + o]; __syncthreads(); }
  float inv = 1.0f / red[0];
  if (j < NP_) {
    float wv = (j < N_) ? (mask[b * NP_ + j] ? (1.0f / (float)N_) : e * inv) : 0.f;
    row[j] = (h16)wv;
  }
}

// ---------------- replace masked token v-rows with per-head mean ------------
__global__ void vmask_kernel(h16* __restrict__ vh, h16* __restrict__ vt,
                             const float* __restrict__ vmh,
                             const unsigned char* __restrict__ mask) {
  int h = blockIdx.x, b = blockIdx.y, d = threadIdx.x;
  h16 mv = (h16)vmh[((size_t)b * H_ + h) * HD_ + d];
  h16* base  = vh + (((size_t)b * H_ + h) * NP_) * HD_ + d;
  h16* baseT = vt + (((size_t)b * H_ + h) * HD_ + d) * NP_;
  const unsigned char* mk = mask + b * NP_;
  for (int n = 0; n < N_; ++n)
    if (mk[n]) { base[(size_t)n * HD_] = mv; baseT[n] = mv; }
}

// ---------------- attn-out: one block per (b,h), V^T-tile in LDS ------------
// vt is d-major [H*B][64][224] so B fragments are K-contiguous.
__global__ void gemm_av_kernel(const h16* __restrict__ wh, const h16* __restrict__ vt,
                               h16* __restrict__ ah) {
  __shared__ __align__(16) h16 Vts[HD_ * NP_];  // 64x224 = 28 KB
  int tid = threadIdx.x, lane = tid & 31, w = tid >> 5;   // w = 0..13
  int bh = blockIdx.x;
  const h16* V = vt + (size_t)bh * HD_ * NP_;
#pragma unroll
  for (int s = tid; s < (HD_ * NP_) / 8; s += 448)
    copy16_g2l(Vts + s * 8, V + (size_t)s * 8);
  wait_async_copies();
  __syncthreads();
  const h16* W = wh + (size_t)bh * NP_ * NP_ + (size_t)w * 16 * NP_;
  v8f acc[4];
#pragma unroll
  for (int t = 0; t < 4; ++t) acc[t] = (v8f){};
#pragma unroll 1
  for (int k0 = 0; k0 < NP_; k0 += 32) {
    v16h a = load_a_rm(W + k0, NP_, lane);
#pragma unroll
    for (int nt = 0; nt < 4; ++nt) {
      v16h b = load_b_tr(Vts + (size_t)(nt * 16) * NP_ + k0, NP_, lane);
      acc[nt] = WMMA_F16(a, b, acc[nt]);
    }
  }
  int col = lane & 15, rb = (lane >> 4) * 8;
  int b_ = bh / H_, h = bh % H_;
#pragma unroll
  for (int nt = 0; nt < 4; ++nt)
#pragma unroll
    for (int i = 0; i < 8; ++i) {
      int n = w * 16 + rb + i;
      if (n < N_)
        ah[((size_t)b_ * N_ + n) * C_ + h * HD_ + nt * 16 + col] = (h16)acc[nt][i];
    }
}

// ---------------- proj GEMM: 128x128 tiles, 8 waves, LDS staged -------------
// B = wpt [768][768] (transposed weight, K contiguous)
__global__ void gemm_proj_kernel(const h16* __restrict__ ah, const h16* __restrict__ wpt,
                                 const float* __restrict__ bias, float* __restrict__ out) {
  __shared__ __align__(16) h16 As[128 * 32];
  __shared__ __align__(16) h16 Bts[128 * 32];
  int tid = threadIdx.x, lane = tid & 31, w = tid >> 5;
  int wm = (w >> 2) * 64, wn = (w & 3) * 32;
  int bm = blockIdx.x, bn = blockIdx.y;
  const h16* Ab = ah + (size_t)bm * 128 * C_;
  const h16* Bb = wpt + (size_t)bn * 128 * C_;
  v8f acc[4][2] = {};
#pragma unroll 1
  for (int k0 = 0; k0 < C_; k0 += 32) {
#pragma unroll
    for (int s = tid; s < 512; s += 256) {
      int row = s >> 2, co = (s & 3) * 8;
      copy16_g2l(As + row * 32 + co, Ab + (size_t)row * C_ + k0 + co);
    }
#pragma unroll
    for (int s = tid; s < 512; s += 256) {
      int row = s >> 2, co = (s & 3) * 8;
      copy16_g2l(Bts + row * 32 + co, Bb + (size_t)row * C_ + k0 + co);
    }
    wait_async_copies();
    __syncthreads();
    v16h bfr[2];
#pragma unroll
    for (int nt = 0; nt < 2; ++nt) bfr[nt] = load_b_tr(Bts + (wn + nt * 16) * 32, 32, lane);
#pragma unroll
    for (int mt = 0; mt < 4; ++mt) {
      v16h a = load_a_rm(As + (wm + mt * 16) * 32, 32, lane);
#pragma unroll
      for (int nt = 0; nt < 2; ++nt) acc[mt][nt] = WMMA_F16(a, bfr[nt], acc[mt][nt]);
    }
    __syncthreads();
  }
  int col = lane & 15, rb = (lane >> 4) * 8;
#pragma unroll
  for (int nt = 0; nt < 2; ++nt) {
    int gj = bn * 128 + wn + nt * 16 + col;
    float bb = bias[gj];
#pragma unroll
    for (int mt = 0; mt < 4; ++mt)
#pragma unroll
      for (int i = 0; i < 8; ++i) {
        int gi = bm * 128 + wm + mt * 16 + rb + i;
        out[(size_t)gi * C_ + gj] = acc[mt][nt][i] + bb;
      }
  }
}

// ---------------------------------------------------------------------------
extern "C" void kernel_launch(void* const* d_in, const int* in_sizes, int n_in,
                              void* d_out, int out_size, void* d_ws, size_t ws_size,
                              hipStream_t stream) {
  const float* x     = (const float*)d_in[0];
  const float* Wqkv  = (const float*)d_in[1];
  const float* Wproj = (const float*)d_in[2];
  const float* bproj = (const float*)d_in[3];
  float* out = (float*)d_out;

  char* ws = (char*)d_ws;
  size_t off = 0;
  auto alloc = [&](size_t bytes) -> void* {
    void* p = ws + off;
    off += (bytes + 255) & ~(size_t)255;
    return p;
  };

  const size_t M = (size_t)B_ * N_;           // 25216 = 197*128
  h16* xh  = (h16*)alloc(M * C_ * 2);
  h16* wqt = (h16*)alloc((size_t)3 * C_ * C_ * 2);   // W_qkv^T [2304][768]
  h16* wpt = (h16*)alloc((size_t)C_ * C_ * 2);       // W_proj^T [768][768]
  h16* qh  = (h16*)alloc((size_t)B_ * H_ * NP_ * HD_ * 2);
  h16* kh  = (h16*)alloc((size_t)B_ * H_ * NP_ * HD_ * 2);
  h16* vh  = (h16*)alloc((size_t)B_ * H_ * NP_ * HD_ * 2);
  h16* vt  = (h16*)alloc((size_t)B_ * H_ * HD_ * NP_ * 2);  // v^T (d-major)
  h16* wh  = (h16*)alloc((size_t)B_ * H_ * NP_ * NP_ * 2);
  h16* ah  = (h16*)alloc(M * C_ * 2);
  float* vmh = (float*)alloc((size_t)B_ * H_ * HD_ * 4);
  float* vmi = (float*)alloc((size_t)B_ * HD_ * 4);
  float* dist = (float*)alloc((size_t)B_ * NPAT * NPAT * 4);
  unsigned char* mask = (unsigned char*)alloc((size_t)B_ * NP_);

  // 1. casts (x straight; weights transposed so GEMM-B is K-contiguous)
  {
    int n0 = (int)(M * C_);
    cast_f16_kernel<<<(n0 + 255) / 256, 256, 0, stream>>>(x, xh, n0);
    int n1 = 3 * C_ * C_;
    castT_f16_kernel<<<(n1 + 255) / 256, 256, 0, stream>>>(Wqkv, wqt, C_, 3 * C_);
    int n2 = C_ * C_;
    castT_f16_kernel<<<(n2 + 255) / 256, 256, 0, stream>>>(Wproj, wpt, C_, C_);
  }

  // 2. zero padded q/k/v (pad rows/cols must be 0)
  size_t qkvBytes = (size_t)B_ * H_ * NP_ * HD_ * 2;
  (void)hipMemsetAsync(qh, 0, qkvBytes, stream);
  (void)hipMemsetAsync(kh, 0, qkvBytes, stream);
  (void)hipMemsetAsync(vh, 0, qkvBytes, stream);
  (void)hipMemsetAsync(vt, 0, qkvBytes, stream);

  // 3. qkv GEMM (scale folded into q): (25216x768)@(768x2304)
  gemm_qkv_kernel<<<dim3((unsigned)(M / 128), 3 * C_ / 128), 256, 0, stream>>>(
      xh, wqt, qh, kh, vh, vt);

  // 4. v statistics -> per-head and global means
  stats_kernel<<<B_, 256, 0, stream>>>(vh, vmh, vmi);

  // 5. patch distances
  dist_kernel<<<B_, NP_, 0, stream>>>(vh, vmi, dist);

  // 6. pool + argmax -> token mask
  mask_kernel<<<B_, 256, 0, stream>>>(dist, mask);

  // 7. attention scores q @ k^T  (one block per (b,h), 14 waves)
  gemm_scores_kernel<<<B_ * H_, 448, 0, stream>>>(qh, kh, wh);

  // 8. softmax + masked-key substitution (1/197)
  softmax_mask_kernel<<<dim3(N_, H_, B_), 256, 0, stream>>>(wh, mask);

  // 9. masked token v-rows -> per-head mean (both copies)
  vmask_kernel<<<dim3(H_, B_), HD_, 0, stream>>>(vh, vt, vmh, mask);

  // 10. attn output w @ v (one block per (b,h), 14 waves)
  gemm_av_kernel<<<B_ * H_, 448, 0, stream>>>(wh, vt, ah);

  // 11. projection + bias -> f32 output: (25216x768)@(768x768)
  gemm_proj_kernel<<<dim3((unsigned)(M / 128), C_ / 128), 256, 0, stream>>>(ah, wpt, bproj, out);
}
